// MOE_14877766713839
// MI455X (gfx1250) — matrix operand order repaired and checked by reading, not compiled, not run on hardware
//
#include <hip/hip_runtime.h>

// ---------------------------------------------------------------------------
// MoE for MI455X (gfx1250, wave32, WMMA).
// Compute-bound (~275 GFLOP vs ~140 MB compulsory traffic) -> all GEMMs on
// v_wmma_f32_16x16x32_bf16 with f32 accumulation, operands pre-packed in the
// exact WMMA fragment layouts.  Key scheduling choice: 128 tokens per block so
// each weight fragment load feeds 8 WMMAs (weight L2 traffic ~2 GB instead of
// 16 GB with 16-token blocks).  Gate weight w_e is folded into h1 (pre-scaled)
// and w_e*b2 enters GEMM2 as an augmented-K chunk, so a single register
// accumulator (128 VGPRs/wave) carries the weighted sum across all 8 experts.
// ---------------------------------------------------------------------------

typedef __attribute__((ext_vector_type(16))) __bf16 v16bf;
typedef __attribute__((ext_vector_type(8)))  float  v8f;

constexpr int N_TOK = 65536, DIM = 256, HID = 512, OUTD = 256, NE = 8, GHID = 128;
constexpr int KC_D = DIM  / 32;   // 8  k-chunks (K=256)
constexpr int KC_H = HID  / 32;   // 16 k-chunks (K=512)
constexpr int NT_H = HID  / 16;   // 32 hidden n-tiles
constexpr int NT_O = OUTD / 16;   // 16 output n-tiles
constexpr int NT_G = GHID / 16;   // 8  gate-hidden n-tiles
constexpr int W2C  = KC_H + 1;    // 17 chunks: 16 of W2 + 1 bias(b2) chunk

// round-to-nearest-even f32 -> bf16 bits
static __device__ __forceinline__ unsigned short f2bf(float f) {
  union { float f; unsigned u; } v; v.f = f;
  unsigned u = v.u + (0x7FFFu + ((v.u >> 16) & 1u));
  return (unsigned short)(u >> 16);
}
static __device__ __forceinline__ v8f vzero() {
  v8f z = {0.f,0.f,0.f,0.f,0.f,0.f,0.f,0.f};
  return z;
}
static __device__ __forceinline__ v16bf ldfrag(const unsigned* p) {
  return *(const v16bf*)p;
}
static __device__ __forceinline__ v8f wmma_bf16(v16bf a, v16bf b, v8f c) {
  return __builtin_amdgcn_wmma_f32_16x16x32_bf16(false, a, false, b, (short)0, c,
                                                 false, false);
}

// --------------------------------------------------------------------------
// Pack x[N,D] f32 -> bf16 A-fragment layout.  Fragment (tile,c) = 256 u32:
// lane L (m=L%16, half=L/16), u32 r holds K = 32c + (r/4)*16 + 8*half + 2*(r%4).
// --------------------------------------------------------------------------
__global__ __launch_bounds__(256) void pack_x_kernel(const float* __restrict__ x,
                                                     unsigned* __restrict__ xp) {
  int idx  = blockIdx.x * 256 + threadIdx.x;       // [0, N*D/2)
  int r    = idx & 7;
  int lane = (idx >> 3) & 31;
  int c    = (idx >> 8) & (KC_D - 1);
  int tile = idx >> 11;
  int m = lane & 15, half = lane >> 4;
  int k = c * 32 + (r >> 2) * 16 + half * 8 + (r & 3) * 2;
  const float* s = x + (tile * 16 + m) * DIM + k;
  xp[idx] = (unsigned)f2bf(s[0]) | ((unsigned)f2bf(s[1]) << 16);
}

// --------------------------------------------------------------------------
// Pack W[mat][Kd][Nd] f32 -> bf16 B-fragment layout.  Fragment (mat,nt,c):
// lane L holds N = 16nt + L%16; u32 r holds K = 32c + (r/4)*16 + 8*(L/16) + 2*(r%4) (+0,+1).
// --------------------------------------------------------------------------
__global__ __launch_bounds__(256) void pack_b_kernel(const float* __restrict__ w,
                                                     unsigned* __restrict__ wp,
                                                     int Kd, int Nd, int total) {
  int idx = blockIdx.x * 256 + threadIdx.x;
  if (idx >= total) return;
  int NC = Kd >> 5, NT = Nd >> 4;
  int r  = idx & 7, L = (idx >> 3) & 31;
  int rest = idx >> 8;
  int c  = rest % NC; rest /= NC;
  int nt = rest % NT;
  int mat = rest / NT;
  int k = c * 32 + (r >> 2) * 16 + (L >> 4) * 8 + (r & 3) * 2, n = nt * 16 + (L & 15);
  const float* s = w + ((long)mat * Kd + k) * Nd + n;
  wp[idx] = (unsigned)f2bf(s[0]) | ((unsigned)f2bf(s[Nd]) << 16);
}

// W2 + b2 packed together: 16 normal chunks plus chunk 16 = bias row
// (B[512][n] = b2[n], other K rows zero) for the augmented-K bias trick.
__global__ __launch_bounds__(256) void pack_w2b_kernel(const float* __restrict__ W2,
                                                       const float* __restrict__ b2,
                                                       unsigned* __restrict__ wp) {
  int idx = blockIdx.x * 256 + threadIdx.x;        // [0, NE*NT_O*W2C*256)
  int r = idx & 7, L = (idx >> 3) & 31;
  int rest = idx >> 8;
  int c  = rest % W2C; rest /= W2C;
  int nt = rest % NT_O;
  int e  = rest / NT_O;
  int n = nt * 16 + (L & 15);
  unsigned val = 0u;
  if (c < KC_H) {
    int k = c * 32 + (r >> 2) * 16 + (L >> 4) * 8 + (r & 3) * 2;
    const float* s = W2 + ((long)e * HID + k) * OUTD + n;
    val = (unsigned)f2bf(s[0]) | ((unsigned)f2bf(s[OUTD]) << 16);
  } else if (L < 16 && r == 0) {
    const float* s = b2 + e * OUTD + n;
    val = (unsigned)f2bf(s[0]);
  }
  wp[idx] = val;
}

// --------------------------------------------------------------------------
// Gating: one wave per 64 tokens (4 tiles) so each Wg1 fragment feeds 4 WMMAs.
//   g = relu(x @ Wg1 + bg1)  -> LDS; logits = g @ Wg2 + bg2 (VALU);
//   top-2 -> 2-way softmax; non-top-k weights exact zero.
// --------------------------------------------------------------------------
__global__ __launch_bounds__(32) void gate_kernel(
    const unsigned* __restrict__ xp, const unsigned* __restrict__ wg1p,
    const float* __restrict__ bg1, const float* __restrict__ Wg2,
    const float* __restrict__ bg2, float* __restrict__ gate_w) {
  constexpr int GP = GHID + 4;
  __shared__ float gbuf[64 * GP];
  __shared__ float lbuf[64 * NE];
  int lane = threadIdx.x, m = lane & 15, half = lane >> 4;
  int tile0 = blockIdx.x * 4;

  for (int nt = 0; nt < NT_G; nt++) {
    v8f hacc[4];
#pragma unroll
    for (int t = 0; t < 4; t++) hacc[t] = vzero();
#pragma unroll
    for (int c = 0; c < KC_D; c++) {
      v16bf b = ldfrag(wg1p + (nt * KC_D + c) * 256 + lane * 8);
#pragma unroll
      for (int t = 0; t < 4; t++) {
        v16bf a = ldfrag(xp + ((tile0 + t) * KC_D + c) * 256 + lane * 8);
        hacc[t] = wmma_bf16(a, b, hacc[t]);
      }
    }
    float bias = bg1[nt * 16 + m];
#pragma unroll
    for (int t = 0; t < 4; t++)
#pragma unroll
      for (int j = 0; j < 8; j++) {
        float v = hacc[t][j] + bias;
        gbuf[(t * 16 + j + 8 * half) * GP + nt * 16 + m] = v > 0.f ? v : 0.f;
      }
  }
  __syncthreads();
  for (int p = 0; p < 16; p++) {
    int id = lane * 16 + p;            // 512 (token, expert) pairs
    int tm = id >> 3, ee = id & 7;
    float s = bg2[ee];
    for (int k2 = 0; k2 < GHID; k2++) s += gbuf[tm * GP + k2] * Wg2[k2 * NE + ee];
    lbuf[tm * NE + ee] = s;
  }
  __syncthreads();
  for (int q = 0; q < 2; q++) {
    int tm = lane * 2 + q;
    float v[NE];
    for (int ee = 0; ee < NE; ee++) v[ee] = lbuf[tm * NE + ee];
    int i1 = 0; float v1 = v[0];
    for (int ee = 1; ee < NE; ee++) if (v[ee] > v1) { v1 = v[ee]; i1 = ee; }
    int i2 = -1; float v2 = -3.4e38f;
    for (int ee = 0; ee < NE; ee++) if (ee != i1 && v[ee] > v2) { v2 = v[ee]; i2 = ee; }
    float e2 = __expf(v2 - v1);
    float den = 1.f + e2;
    float w1 = 1.f / den, w2 = e2 / den;
    float* gw = gate_w + ((long)(tile0 * 16 + tm)) * NE;
    for (int ee = 0; ee < NE; ee++) gw[ee] = (ee == i1) ? w1 : (ee == i2 ? w2 : 0.f);
  }
}

// --------------------------------------------------------------------------
// Experts: 8 waves per 128 tokens, all 8 experts accumulated into one register
// output tile (oacc: 8 token tiles x 2 O n-tiles x v8f = 128 VGPRs/wave).
// Per expert, per hidden half (256 wide):
//   GEMM1: h = relu(x@W1+b1) * w_e  -> bf16 in 64KB XOR-swizzled LDS
//   GEMM2: oacc += h_lds @ W2 (8 chunks), B frags reused across 8 token tiles
// then one augmented-K WMMA adds w_e*b2.
// --------------------------------------------------------------------------
__global__ __launch_bounds__(256) void expert_kernel(
    const unsigned* __restrict__ xp, const unsigned* __restrict__ w1p,
    const unsigned* __restrict__ w2p, const float* __restrict__ b1,
    const float* __restrict__ gate_w, float* __restrict__ out) {
  __shared__ unsigned h1[128 * 128];   // 64KB: [token row][u32 col ^ (m<<2)]
  int tid = threadIdx.x;
  int wave = tid >> 5, lane = tid & 31, m = lane & 15, half = lane >> 4;
  int tile0 = blockIdx.x * 8;
  long tokbase = (long)tile0 * 16;

  v8f oacc[8][2];
#pragma unroll
  for (int t = 0; t < 8; t++) { oacc[t][0] = vzero(); oacc[t][1] = vzero(); }

  for (int e = 0; e < NE; e++) {
    const unsigned* w1e = w1p + e * (NT_H * KC_D * 256);
    const unsigned* w2e = w2p + e * (NT_O * W2C * 256);

    for (int h = 0; h < 2; h++) {
      __syncthreads();            // previous readers of h1 are done
      // ---- GEMM1: wave owns hidden n-tiles nt = 16h + 2*wave + p ----
#pragma unroll
      for (int p = 0; p < 2; p++) {
        int nt = h * 16 + wave * 2 + p;
        v8f hacc[8];
#pragma unroll
        for (int t = 0; t < 8; t++) hacc[t] = vzero();
#pragma unroll
        for (int c = 0; c < KC_D; c++) {
          v16bf b = ldfrag(w1e + (nt * KC_D + c) * 256 + lane * 8);
#pragma unroll
          for (int t = 0; t < 8; t++) {
            v16bf a = ldfrag(xp + ((tile0 + t) * KC_D + c) * 256 + lane * 8);
            hacc[t] = wmma_bf16(a, b, hacc[t]);
          }
        }
        float bias = b1[e * HID + nt * 16 + m];
        int ncol = (wave * 2 + p) * 16 + m;        // local col in half, 0..255
#pragma unroll
        for (int t = 0; t < 8; t++) {
#pragma unroll
          for (int j = 0; j < 8; j++) {
            int row = t * 16 + j + 8 * half;       // token row 0..127
            float wgt = gate_w[(tokbase + row) * NE + e];
            float v = hacc[t][j] + bias;
            v = v > 0.f ? v : 0.f;
            unsigned col32 = ((unsigned)(ncol >> 1)) ^ (((unsigned)row & 15u) << 2);
            ((unsigned short*)h1)[(row * 128 + col32) * 2 + (ncol & 1)] = f2bf(v * wgt);
          }
        }
      }
      __syncthreads();
      // ---- GEMM2: oacc += h1(half) @ W2 chunks c2 = 8h..8h+7 ----
#pragma unroll
      for (int cl = 0; cl < 8; cl++) {
        int c2 = h * 8 + cl;
        v16bf bfrag[2];
#pragma unroll
        for (int i = 0; i < 2; i++)
          bfrag[i] = ldfrag(w2e + ((wave * 2 + i) * W2C + c2) * 256 + lane * 8);
#pragma unroll
        for (int t = 0; t < 8; t++) {
          union { v16bf v; unsigned u[8]; } af;
          int row = t * 16 + m;
          unsigned rb = (unsigned)row * 128u;
          unsigned sw = ((unsigned)m) << 2;
#pragma unroll
          for (int r = 0; r < 8; r++) {
            int kl = cl * 32 + (r >> 2) * 16 + half * 8 + (r & 3) * 2;
            af.u[r] = h1[rb + (((unsigned)kl >> 1) ^ sw)];
          }
#pragma unroll
          for (int i = 0; i < 2; i++)
            oacc[t][i] = wmma_bf16(af.v, bfrag[i], oacc[t][i]);
        }
      }
    }
    // ---- augmented-K bias chunk: oacc += w_e * b2 ----
    {
      v16bf bb[2];
#pragma unroll
      for (int i = 0; i < 2; i++)
        bb[i] = ldfrag(w2e + ((wave * 2 + i) * W2C + KC_H) * 256 + lane * 8);
#pragma unroll
      for (int t = 0; t < 8; t++) {
        union { v16bf v; unsigned u[8]; } af;
#pragma unroll
        for (int r = 0; r < 8; r++) af.u[r] = 0u;
        if (half == 0) {
          float wv = gate_w[(tokbase + t * 16 + m) * NE + e];
          af.u[0] = (unsigned)f2bf(wv);            // A[m][K=512] = w_e
        }
#pragma unroll
        for (int i = 0; i < 2; i++)
          oacc[t][i] = wmma_bf16(af.v, bb[i], oacc[t][i]);
      }
    }
    // prefetch next expert's packed weights into GL2 (global_prefetch_b8)
    if (e + 1 < NE) {
      __builtin_prefetch(w1p + (e + 1) * (NT_H * KC_D * 256), 0, 1);
      __builtin_prefetch(w2p + (e + 1) * (NT_O * W2C * 256), 0, 1);
    }
  }
  // ---- store 128x256 output tile ----
#pragma unroll
  for (int t = 0; t < 8; t++)
#pragma unroll
    for (int i = 0; i < 2; i++) {
      int n = (wave * 2 + i) * 16 + m;
#pragma unroll
      for (int j = 0; j < 8; j++)
        out[(tokbase + t * 16 + j + 8 * half) * OUTD + n] = oacc[t][i][j];
    }
}

extern "C" void kernel_launch(void* const* d_in, const int* in_sizes, int n_in,
                              void* d_out, int out_size, void* d_ws, size_t ws_size,
                              hipStream_t stream) {
  const float* x   = (const float*)d_in[0];
  const float* Wg1 = (const float*)d_in[1];
  const float* bg1 = (const float*)d_in[2];
  const float* Wg2 = (const float*)d_in[3];
  const float* bg2 = (const float*)d_in[4];
  const float* W1  = (const float*)d_in[5];
  const float* b1  = (const float*)d_in[6];
  const float* W2  = (const float*)d_in[7];
  const float* b2  = (const float*)d_in[8];
  float* out = (float*)d_out;

  char* ws = (char*)d_ws;
  size_t off = 0;
  unsigned* xp   = (unsigned*)(ws + off); off += (size_t)N_TOK * DIM / 2 * 4;        // 32 MB
  unsigned* w1p  = (unsigned*)(ws + off); off += (size_t)NE * DIM * HID / 2 * 4;     // 2 MB
  unsigned* w2p  = (unsigned*)(ws + off); off += (size_t)NE * NT_O * W2C * 256 * 4;  // 2.125 MB
  unsigned* wg1p = (unsigned*)(ws + off); off += (size_t)DIM * GHID / 2 * 4;         // 64 KB
  float* gate_w  = (float*)(ws + off);    off += (size_t)N_TOK * NE * 4;             // 2 MB

  pack_x_kernel<<<N_TOK * DIM / 2 / 256, 256, 0, stream>>>(x, xp);
  {
    int tot = NE * DIM * HID / 2;
    pack_b_kernel<<<(tot + 255) / 256, 256, 0, stream>>>(W1, w1p, DIM, HID, tot);
  }
  {
    int tot = NE * NT_O * W2C * 256;
    pack_w2b_kernel<<<(tot + 255) / 256, 256, 0, stream>>>(W2, b2, w2p);
  }
  {
    int tot = DIM * GHID / 2;
    pack_b_kernel<<<(tot + 255) / 256, 256, 0, stream>>>(Wg1, wg1p, DIM, GHID, tot);
  }

  gate_kernel<<<N_TOK / 64, 32, 0, stream>>>(xp, wg1p, bg1, Wg2, bg2, gate_w);
  expert_kernel<<<N_TOK / 128, 256, 0, stream>>>(xp, w1p, w2p, b1, gate_w, out);
}
